// DynamicNTKLayer_68487548502822
// MI455X (gfx1250) — matrix operand order repaired and checked by reading, not compiled, not run on hardware
//
#include <hip/hip_runtime.h>
#include <hip/hip_bf16.h>

// ---------- CDNA5 WMMA types ----------
typedef __attribute__((ext_vector_type(16))) __bf16 v16bf;
typedef __attribute__((ext_vector_type(8)))  float  v8f;

union FragAB {
  v16bf v;
  uint4 q[2];
  unsigned short u[16];
};

__device__ __forceinline__ unsigned short f32_to_bf16_rne(float f) {
  union { float f; unsigned u; } c; c.f = f;
  unsigned u = c.u;
  unsigned r = (u + 0x7FFFu + ((u >> 16) & 1u)) >> 16;
  return (unsigned short)r;
}
__device__ __forceinline__ float bf16_to_f32(unsigned short h) {
  union { unsigned u; float f; } c; c.u = ((unsigned)h) << 16;
  return c.f;
}

// ---------- f32 -> bf16 conversion ----------
__global__ void k_cvt_f32_bf16(const float* __restrict__ src,
                               unsigned short* __restrict__ dst, long long n) {
  long long i = (long long)blockIdx.x * blockDim.x + threadIdx.x;
  long long stride = (long long)gridDim.x * blockDim.x;
  for (; i < n; i += stride) dst[i] = f32_to_bf16_rne(src[i]);
}

// ---------- NT GEMM: out[m,n] = scale * sum_k A[m,k]*Bt[n,k] (+bias[n]) (+res[m,n]) ----------
// A: (M,K) bf16 row-major, Bt: (N,K) bf16 row-major. 128 threads = 4 waves,
// each wave owns a 32x32 tile of a 64x64 block tile. Pure WMMA-native global loads.
__global__ void __launch_bounds__(128)
k_gemm_nt(const unsigned short* __restrict__ A, const unsigned short* __restrict__ B,
          const float* __restrict__ bias, const float* __restrict__ res,
          float scale, float* __restrict__ outF, unsigned short* __restrict__ outH,
          int M, int N, int K,
          long long sA, long long sB, long long sOut, long long sRes) {
  const int lane = threadIdx.x & 31;
  const int wave = threadIdx.x >> 5;
  const int z = blockIdx.z;
  A += (long long)z * sA;
  B += (long long)z * sB;
  const int m0 = blockIdx.x * 64 + (wave >> 1) * 32;
  const int n0 = blockIdx.y * 64 + (wave & 1) * 32;
  const int row = lane & 15;
  const int hi  = lane >> 4;

  v8f acc00 = {}, acc01 = {}, acc10 = {}, acc11 = {};

  // A fragment (16x32 bf16): lanes 0-15 elems 0-7 = K0-7, 8-15 = K16-23;
  // lanes 16-31 shifted by 8. B fragment (32x16): same K layout as A.
  const unsigned short* pa0 = A + (long long)(m0 + row) * K + hi * 8;
  const unsigned short* pa1 = pa0 + (long long)16 * K;
  const unsigned short* pb0 = B + (long long)(n0 + row) * K + hi * 8;
  const unsigned short* pb1 = pb0 + (long long)16 * K;

  for (int k = 0; k < K; k += 32) {
    FragAB a0, a1, b0, b1;
    a0.q[0] = *(const uint4*)(pa0 + k);
    a0.q[1] = *(const uint4*)(pa0 + k + 16);
    a1.q[0] = *(const uint4*)(pa1 + k);
    a1.q[1] = *(const uint4*)(pa1 + k + 16);
    b0.q[0] = *(const uint4*)(pb0 + k);
    b0.q[1] = *(const uint4*)(pb0 + k + 16);
    b1.q[0] = *(const uint4*)(pb1 + k);
    b1.q[1] = *(const uint4*)(pb1 + k + 16);
    acc00 = __builtin_amdgcn_wmma_f32_16x16x32_bf16(false, a0.v, false, b0.v, (short)0, acc00, false, false);
    acc01 = __builtin_amdgcn_wmma_f32_16x16x32_bf16(false, a0.v, false, b1.v, (short)0, acc01, false, false);
    acc10 = __builtin_amdgcn_wmma_f32_16x16x32_bf16(false, a1.v, false, b0.v, (short)0, acc10, false, false);
    acc11 = __builtin_amdgcn_wmma_f32_16x16x32_bf16(false, a1.v, false, b1.v, (short)0, acc11, false, false);
  }

  // Epilogue: C/D layout — VGPR r holds M = r + 8*(lane>=16), N = lane&15.
  v8f accs[2][2] = { { acc00, acc01 }, { acc10, acc11 } };
  #pragma unroll
  for (int tm = 0; tm < 2; ++tm) {
    #pragma unroll
    for (int tn = 0; tn < 2; ++tn) {
      #pragma unroll
      for (int r = 0; r < 8; ++r) {
        int m = m0 + tm * 16 + r + hi * 8;
        int n = n0 + tn * 16 + row;
        float v = accs[tm][tn][r] * scale;
        if (bias) v += bias[n];
        if (res)  v += res[(long long)z * sRes + (long long)m * N + n];
        long long o = (long long)z * sOut + (long long)m * N + n;
        if (outF) outF[o] = v;
        else      outH[o] = f32_to_bf16_rne(v);
      }
    }
  }
}

// ---------- TN GEMM (LDS-staged): out[m,n] = sum_j A[j,m]*B[j,n], bf16 out ----------
// A: (Kj,M) bf16 (att), B: (Kj,N) bf16 (phi). One wave per 16x16 output tile.
__global__ void __launch_bounds__(32)
k_gemm_tn_bf16(const unsigned short* __restrict__ A, const unsigned short* __restrict__ B,
               unsigned short* __restrict__ outH,
               int M, int N, int Kj,
               long long sA, long long sB, long long sOut) {
  __shared__ __align__(16) unsigned short sa[32 * 16];
  __shared__ __align__(16) unsigned short sb[32 * 16];
  const int lane = threadIdx.x;
  const int z = blockIdx.z;
  A += (long long)z * sA;
  B += (long long)z * sB;
  const int m0 = blockIdx.x * 16;
  const int n0 = blockIdx.y * 16;
  const int ml = lane & 15;
  const int hi = lane >> 4;

  v8f acc = {};
  for (int j = 0; j < Kj; j += 32) {
    const unsigned short* ga = A + (long long)(j + lane) * M + m0;
    const unsigned short* gb = B + (long long)(j + lane) * N + n0;
    uint4 qa0 = *(const uint4*)(ga);
    uint4 qa1 = *(const uint4*)(ga + 8);
    uint4 qb0 = *(const uint4*)(gb);
    uint4 qb1 = *(const uint4*)(gb + 8);
    __syncthreads();
    *(uint4*)(sa + lane * 16)     = qa0;
    *(uint4*)(sa + lane * 16 + 8) = qa1;
    *(uint4*)(sb + lane * 16)     = qb0;
    *(uint4*)(sb + lane * 16 + 8) = qb1;
    __syncthreads();
    FragAB a, b;
    #pragma unroll
    for (int e = 0; e < 16; ++e) {
      int ka = (e < 8 ? e : e + 8) + hi * 8;   // A interleaved K layout
      int kb = (e < 8 ? e : e + 8) + hi * 8;   // B interleaved K layout
      a.u[e] = sa[ka * 16 + ml];
      b.u[e] = sb[kb * 16 + ml];
    }
    acc = __builtin_amdgcn_wmma_f32_16x16x32_bf16(false, a.v, false, b.v, (short)0, acc, false, false);
  }
  #pragma unroll
  for (int r = 0; r < 8; ++r) {
    int m = m0 + r + hi * 8;
    int n = n0 + ml;
    outH[(long long)z * sOut + (long long)m * N + n] = f32_to_bf16_rne(acc[r]);
  }
}

// ---------- tiny 4-wide attention: one thread per (n, h, l) ----------
__global__ void k_attn(const unsigned short* __restrict__ qkv,
                       unsigned short* __restrict__ ao, int Nn) {
  const int HD = 64, L = 4, E = 1024, E3 = 3072;
  int idx = blockIdx.x * blockDim.x + threadIdx.x;
  if (idx >= Nn * 16 * L) return;
  int l = idx & 3;
  int h = (idx >> 2) & 15;
  int n = idx >> 6;

  const unsigned short* qrow = qkv + ((long long)l * Nn + n) * E3 + h * HD;
  float q[64];
  #pragma unroll
  for (int d = 0; d < HD; ++d) q[d] = bf16_to_f32(qrow[d]);

  float s[4];
  const float scale = 0.125f;  // 1/sqrt(64)
  #pragma unroll
  for (int m = 0; m < L; ++m) {
    const unsigned short* krow = qkv + ((long long)m * Nn + n) * E3 + E + h * HD;
    float a = 0.f;
    for (int d = 0; d < HD; ++d) a += q[d] * bf16_to_f32(krow[d]);
    s[m] = a * scale;
  }
  float mx = fmaxf(fmaxf(s[0], s[1]), fmaxf(s[2], s[3]));
  float w[4], sum = 0.f;
  #pragma unroll
  for (int m = 0; m < L; ++m) { w[m] = __expf(s[m] - mx); sum += w[m]; }
  float inv = 1.f / sum;

  unsigned short* orow = ao + ((long long)l * Nn + n) * E + h * HD;
  for (int d = 0; d < HD; ++d) {
    float a = 0.f;
    #pragma unroll
    for (int m = 0; m < L; ++m) {
      const unsigned short* vrow = qkv + ((long long)m * Nn + n) * E3 + 2 * E + h * HD;
      a += w[m] * bf16_to_f32(vrow[d]);
    }
    orow[d] = f32_to_bf16_rne(a * inv);
  }
}

extern "C" void kernel_launch(void* const* d_in, const int* in_sizes, int n_in,
                              void* d_out, int out_size, void* d_ws, size_t ws_size,
                              hipStream_t stream) {
  (void)in_sizes; (void)n_in; (void)out_size; (void)ws_size;
  const float* x   = (const float*)d_in[0];
  const float* fmw = (const float*)d_in[1];
  const float* fmb = (const float*)d_in[2];
  const float* ipw = (const float*)d_in[3];
  const float* ipb = (const float*)d_in[4];
  const float* ow  = (const float*)d_in[5];
  const float* ob  = (const float*)d_in[6];
  float* out = (float*)d_out;

  const int B = 4, N = 4096, D = 1024, E3 = 3072;
  const long long MN = (long long)B * N;  // 16384 flattened rows

  char* ws = (char*)d_ws;
  size_t off = 0;
  auto alloc = [&](size_t bytes) -> char* {
    char* p = ws + off;
    off += (bytes + 255) & ~(size_t)255;
    return p;
  };
  unsigned short* xb   = (unsigned short*)alloc((size_t)MN * D * 2);      // x bf16
  unsigned short* fmwb = (unsigned short*)alloc((size_t)D * D * 2);       // fm_w bf16
  unsigned short* ipwb = (unsigned short*)alloc((size_t)E3 * D * 2);      // in_proj_w bf16
  unsigned short* owb  = (unsigned short*)alloc((size_t)D * D * 2);       // out_w bf16
  unsigned short* phib = (unsigned short*)alloc((size_t)MN * D * 2);      // phi bf16
  unsigned short* qkvb = (unsigned short*)alloc((size_t)MN * E3 * 2);     // qkv bf16
  unsigned short* aob  = (unsigned short*)alloc((size_t)MN * D * 2);      // attention (pre out-proj)
  unsigned short* attb = (unsigned short*)alloc((size_t)MN * D * 2);      // attended (post out-proj)
  unsigned short* gtb  = (unsigned short*)alloc((size_t)B * D * D * 2);   // Gt[b] = (phi^T att)^T

  // 0) bf16 conversions
  k_cvt_f32_bf16<<<4096, 256, 0, stream>>>(x,   xb,   MN * D);
  k_cvt_f32_bf16<<<1024, 256, 0, stream>>>(fmw, fmwb, (long long)D * D);
  k_cvt_f32_bf16<<<2048, 256, 0, stream>>>(ipw, ipwb, (long long)E3 * D);
  k_cvt_f32_bf16<<<1024, 256, 0, stream>>>(ow,  owb,  (long long)D * D);

  // 1) phi = x @ fm_w^T + fm_b
  {
    dim3 g((unsigned)(MN / 64), D / 64, 1);
    k_gemm_nt<<<g, 128, 0, stream>>>(xb, fmwb, fmb, nullptr, 1.0f, nullptr, phib,
                                     (int)MN, D, D, 0, 0, 0, 0);
  }
  // 2) qkv = x @ in_proj_w^T + in_proj_b
  {
    dim3 g((unsigned)(MN / 64), E3 / 64, 1);
    k_gemm_nt<<<g, 128, 0, stream>>>(xb, ipwb, ipb, nullptr, 1.0f, nullptr, qkvb,
                                     (int)MN, E3, D, 0, 0, 0, 0);
  }
  // 3) tiny attention over L=4
  {
    int total = N * 16 * 4;
    k_attn<<<(total + 255) / 256, 256, 0, stream>>>(qkvb, aob, N);
  }
  // 3b) out projection: att = ao @ out_w^T + out_b
  {
    dim3 g((unsigned)(MN / 64), D / 64, 1);
    k_gemm_nt<<<g, 128, 0, stream>>>(aob, owb, ob, nullptr, 1.0f, nullptr, attb,
                                     (int)MN, D, D, 0, 0, 0, 0);
  }
  // 4) Gt[b][n,k] = sum_j att[b,j,n] * phi[b,j,k]   (avoids the 256MB NxN kernel matrix)
  {
    dim3 g(D / 16, D / 16, B);
    k_gemm_tn_bf16<<<g, 32, 0, stream>>>(attb, phib, gtb, D, D, N,
                                         (long long)N * D, (long long)N * D,
                                         (long long)D * D);
  }
  // 5) out = x + 0.5 * phi[b] @ Gt[b]^T  (NT with Gt rows as K-vectors)
  {
    dim3 g(N / 64, D / 64, B);
    k_gemm_nt<<<g, 128, 0, stream>>>(phib, gtb, nullptr, x, 0.5f, out, nullptr,
                                     N, D, D,
                                     (long long)N * D, (long long)D * D,
                                     (long long)N * D, (long long)N * D);
  }
}
